// ConditionalRandomField_42468636623419
// MI455X (gfx1250) — matrix operand order repaired and checked by reading, not compiled, not run on hardware
//
#include <hip/hip_runtime.h>
#include <hip/hip_bf16.h>
#include <hip/hip_fp16.h>

typedef __attribute__((ext_vector_type(16))) _Float16 v16h;
typedef __attribute__((ext_vector_type(8)))  _Float16 v8h;
typedef __attribute__((ext_vector_type(8)))  float    v8f;

#define B_ 256
#define L_ 512
#define T_ 128
#define AP_ 132   // padded alpha row stride (floats) to spread LDS banks

// ---------------------------------------------------------------------------
// Precompute column maxes c[j] = max_i trans[i][j] and E[i][j] = exp(trans-c)
// in f16. Runs once; E is reused by every scan step (step-invariant B matrix).
// ---------------------------------------------------------------------------
__global__ void crf_init_E(const float* __restrict__ trans,
                           _Float16* __restrict__ E,
                           float* __restrict__ c) {
  int j = threadIdx.x;            // 128 threads, one per column
  float cm = -INFINITY;
  for (int i = 0; i < T_; ++i) cm = fmaxf(cm, trans[i * T_ + j]);
  c[j] = cm;
  for (int i = 0; i < T_; ++i)
    E[i * T_ + j] = (_Float16)__expf(trans[i * T_ + j] - cm);
}

// ---------------------------------------------------------------------------
// Forward algorithm: one workgroup handles a 16-batch tile for all 512 steps.
// 8 waves; wave w owns output columns [16w, 16w+16). Per step: max-shift ->
// exp -> f16 A in LDS -> 4x v_wmma_f32_16x16x32_f16 (K=128) -> log epilogue.
// Writes logsumexp(alpha) (the normalizer) into out[b].
// ---------------------------------------------------------------------------
__global__ __launch_bounds__(256) void crf_forward(
    const float* __restrict__ feats,   // [B, L, T]
    const int*   __restrict__ mask,    // [B, L]
    const _Float16* __restrict__ E,    // [T, T] exp(trans - c_j)
    const float* __restrict__ c,       // [T]
    float* __restrict__ out)           // [B] normalizer
{
  __shared__ __align__(16) float    s_alpha[16 * AP_];
  __shared__ __align__(16) _Float16 s_A[16 * T_];     // exp(alpha - m_b), f16
  __shared__ float s_m[16];

  const int tid   = threadIdx.x;
  const int lane  = tid & 31;
  const int wave  = tid >> 5;          // 0..7 -> N tile
  const int bTile = blockIdx.x;        // 0..15
  const int b0    = bTile * 16;

  const int n0  = wave * 16;           // this wave's output column base
  const int ln  = lane & 15;
  const int sel = lane >> 4;           // half-wave selector

  // ---- preload B fragments of E (32x16 f16 each, contiguous K, same order as A):
  //      lanes 0-15 hold K=kk..kk+15, lanes 16-31 hold K=kk+16..kk+31 ----
  v16h bfrag[4];
  for (int kk = 0; kk < 4; ++kk) {
    const int kbase = kk * 32 + sel * 16;
    for (int q = 0; q < 16; ++q)
      bfrag[kk][q] = E[(kbase + q) * T_ + (n0 + ln)];
  }
  const float cj = c[n0 + ln];

  // ---- reduction mapping: 16 threads per batch row, 8 contiguous j each ----
  const int rb = tid >> 4;             // batch row 0..15
  const int jl = tid & 15;

  // alpha0 = feats[:, 0, :]
  for (int k = 0; k < 8; ++k) {
    int j = jl * 8 + k;
    s_alpha[rb * AP_ + j] = feats[(size_t)(b0 + rb) * L_ * T_ + j];
  }
  __syncthreads();

  for (int t = 1; t < L_; ++t) {
    // prefetch this step's emissions (consumed in the epilogue below)
    __builtin_prefetch(&feats[(size_t)(b0 + rb) * L_ * T_ + (size_t)t * T_ + jl * 8], 0, 0);

    // ---- phase 1: per-batch max, then A = exp(alpha - m_b) as f16 ----
    float vals[8];
    float lm = -INFINITY;
    for (int k = 0; k < 8; ++k) {
      vals[k] = s_alpha[rb * AP_ + jl * 8 + k];
      lm = fmaxf(lm, vals[k]);
    }
    for (int off = 1; off < 16; off <<= 1)
      lm = fmaxf(lm, __shfl_xor(lm, off, 16));
    if (jl == 0) s_m[rb] = lm;
    for (int k = 0; k < 8; ++k)
      s_A[rb * T_ + jl * 8 + k] = (_Float16)__expf(vals[k] - lm);
    __syncthreads();

    // ---- phase 2: acc(16x16) = A(16x128) * E(128x16) via 4 WMMAs ----
    v8f acc = {};
    const int arow = lane & 15;        // A: M = lane%16
    for (int kk = 0; kk < 4; ++kk) {
      const int abase = kk * 32 + sel * 16;  // lanes 0-15: K base 0; 16-31: +16
      v8h lo = *(const v8h*)&s_A[arow * T_ + abase];        // K = abase..+7
      v8h hi = *(const v8h*)&s_A[arow * T_ + abase + 8];    // K = abase+8..+15
      v16h afrag;
      for (int q = 0; q < 8; ++q) { afrag[q] = lo[q]; afrag[q + 8] = hi[q]; }
      acc = __builtin_amdgcn_wmma_f32_16x16x32_f16(
          /*neg_a=*/false, afrag, /*neg_b=*/false, bfrag[kk],
          /*c_mod=*/(short)0, acc, /*reuse_a=*/false, /*reuse_b=*/false);
    }

    // ---- phase 3: epilogue. C layout: VGPR r -> M = r + 8*sel, N = lane%16
    #pragma unroll
    for (int r = 0; r < 8; ++r) {
      const int M  = r + sel * 8;
      const int j  = n0 + ln;
      const int gb = b0 + M;
      float oldv = s_alpha[M * AP_ + j];
      float mt   = (float)mask[(size_t)gb * L_ + t];
      float emit = feats[(size_t)gb * L_ * T_ + (size_t)t * T_ + j];
      float nv   = __logf(acc[r]) + s_m[M] + cj + emit;
      s_alpha[M * AP_ + j] = (mt > 0.f) ? nv : oldv;
    }
    __syncthreads();
  }

  // ---- final normalizer: logsumexp over states ----
  float vals[8];
  float lm = -INFINITY;
  for (int k = 0; k < 8; ++k) {
    vals[k] = s_alpha[rb * AP_ + jl * 8 + k];
    lm = fmaxf(lm, vals[k]);
  }
  for (int off = 1; off < 16; off <<= 1)
    lm = fmaxf(lm, __shfl_xor(lm, off, 16));
  float s = 0.f;
  for (int k = 0; k < 8; ++k) s += __expf(vals[k] - lm);
  for (int off = 1; off < 16; off <<= 1)
    s += __shfl_xor(s, off, 16);
  if (jl == 0) out[b0 + rb] = lm + __logf(s);
}

// ---------------------------------------------------------------------------
// Gold path score: out[b] -= sum_t emit[b,t,tag]*mask + sum_{t>=1} trans*mask
// ---------------------------------------------------------------------------
__global__ void crf_gold(const float* __restrict__ feats,
                         const int*   __restrict__ tags,
                         const int*   __restrict__ mask,
                         const float* __restrict__ trans,
                         float* __restrict__ out) {
  int b = blockIdx.x * blockDim.x + threadIdx.x;
  if (b >= B_) return;
  float g = 0.f;
  int prev = tags[(size_t)b * L_];
  g += feats[(size_t)b * L_ * T_ + prev] * (float)mask[(size_t)b * L_];
  for (int t = 1; t < L_; ++t) {
    int cur  = tags[(size_t)b * L_ + t];
    float mt = (float)mask[(size_t)b * L_ + t];
    g += (trans[prev * T_ + cur] +
          feats[(size_t)b * L_ * T_ + (size_t)t * T_ + cur]) * mt;
    prev = cur;
  }
  out[b] -= g;
}

extern "C" void kernel_launch(void* const* d_in, const int* in_sizes, int n_in,
                              void* d_out, int out_size, void* d_ws, size_t ws_size,
                              hipStream_t stream) {
  (void)in_sizes; (void)n_in; (void)out_size; (void)ws_size;
  const float* feats = (const float*)d_in[0];
  const int*   tags  = (const int*)d_in[1];
  const int*   mask  = (const int*)d_in[2];
  const float* trans = (const float*)d_in[3];
  float* out = (float*)d_out;

  _Float16* E = (_Float16*)d_ws;                                    // 32 KB
  float*    c = (float*)((char*)d_ws + (size_t)T_ * T_ * sizeof(_Float16));

  crf_init_E<<<1, T_, 0, stream>>>(trans, E, c);
  crf_forward<<<B_ / 16, 256, 0, stream>>>(feats, mask, E, c, out);
  crf_gold<<<1, 256, 0, stream>>>(feats, tags, mask, trans, out);
}
